// HeteroGATModel_71622874628354
// MI455X (gfx1250) — matrix operand order repaired and checked by reading, not compiled, not run on hardware
//
#include <hip/hip_runtime.h>
#include <math.h>
#include <stdint.h>

#define HID 256
#define HEADS 4
#define CPH 64
#define NEG_SLOPE 0.2f
#define IN_DIM_ 128

#define BM 128
#define BN 64
#define BK 32

typedef __attribute__((ext_vector_type(16))) _Float16 v16h;
typedef __attribute__((ext_vector_type(8)))  _Float16 v8h;
typedef __attribute__((ext_vector_type(8)))  float    v8f;

// ---------- helpers ----------
__device__ __forceinline__ unsigned f32key(float f) {
  unsigned u = __float_as_uint(f);
  return (u & 0x80000000u) ? ~u : (u | 0x80000000u);
}
__device__ __forceinline__ float keyf32(unsigned k) {
  unsigned u = (k & 0x80000000u) ? (k & 0x7FFFFFFFu) : ~k;
  return __uint_as_float(u);
}
__device__ __forceinline__ float leaky(float x) { return x > 0.f ? x : NEG_SLOPE * x; }

// ---------- zero fill ----------
__global__ void k_zero(float* __restrict__ p, long long n) {
  long long i = (long long)blockIdx.x * blockDim.x + threadIdx.x;
  if (i < n) p[i] = 0.f;
}

// ---------- LDS-tiled, double-buffered WMMA GEMM ----------
// C[M,N](f16, optional) = A[M,K] @ B[K,N](f32 weights); AT = float or _Float16.
// Optional bias+relu epilogue; optional fused per-head attention dot
// attE[row,h] = sum_col acc[row,col]*attV[col] (BN == CPH: block nb == head nb).
// Block: 256 thr = 8 waves; block tile 128x64; wave tile 16x64 (4 wmma acc).
// Software pipeline: stage k+1 (f16-A via async DMA global->LDS, B via
// register-held float4 loads) overlaps the WMMA block of stage k.
template <typename AT>
__global__ __launch_bounds__(256) void k_wmma_gemm(
    const AT* __restrict__ A, const float* __restrict__ B,
    _Float16* __restrict__ C, const float* __restrict__ bias,
    const float* __restrict__ attV, float* __restrict__ attE,
    int M, int K, int N, int relu)
{
  __shared__ __align__(16) _Float16 As[2][BM][BK];    // 16 KB
  __shared__ __align__(16) _Float16 Bts[2][BN][BK];   // 8 KB  (B transposed: [n][k])

  int tid = threadIdx.x;
  int wv = tid >> 5, lane = tid & 31;
  int l15 = lane & 15, lhalf = (lane >> 4) & 1;

  int nblocks = N / BN;
  int mb = blockIdx.x / nblocks, nb = blockIdx.x % nblocks;
  int m0 = mb * BM, n0 = nb * BN;
  const int nk = K / BK;

  v8f acc[4] = {{}, {}, {}, {}};
  float4 aReg[2];        // f32-A staging registers
  float4 bReg[2];        // B staging registers

  // ---- staging helpers (indices are k0-invariant per thread) ----
  auto issueA = [&](int buf, int k0) {
    if constexpr (sizeof(AT) == 2) {
      // f16 source: async DMA straight into LDS (ASYNCcnt), 2x b128 per thread
#pragma unroll
      for (int it = 0; it < 2; ++it) {
        int f = tid * 2 + it;                     // 0..511
        int row = f >> 2, c8 = (f & 3) << 3;
        int grow = m0 + row; if (grow >= M) grow = M - 1;
        const AT* gp = A + (size_t)grow * K + k0 + c8;
        unsigned lds = (unsigned)(uintptr_t)&As[buf][row][c8];
        asm volatile("global_load_async_to_lds_b128 %0, %1, off"
                     :: "v"(lds), "v"(gp) : "memory");
      }
    } else {
#pragma unroll
      for (int it = 0; it < 2; ++it) {
        int f = tid * 2 + it;                     // 0..511
        int row = f >> 3, c4 = (f & 7) << 2;
        int grow = m0 + row; if (grow >= M) grow = M - 1;
        aReg[it] = *(const float4*)((const float*)A + (size_t)grow * K + k0 + c4);
      }
    }
  };
  auto commitA = [&](int buf) {
    if constexpr (sizeof(AT) != 2) {
#pragma unroll
      for (int it = 0; it < 2; ++it) {
        int f = tid * 2 + it;
        int row = f >> 3, c4 = (f & 7) << 2;
        _Float16* d = &As[buf][row][c4];
        d[0] = (_Float16)aReg[it].x; d[1] = (_Float16)aReg[it].y;
        d[2] = (_Float16)aReg[it].z; d[3] = (_Float16)aReg[it].w;
      }
    }
  };
  auto waitA = [&]() {
    if constexpr (sizeof(AT) == 2)
      asm volatile("s_wait_asynccnt 0" ::: "memory");
  };
  auto issueB = [&](int k0) {
#pragma unroll
    for (int it = 0; it < 2; ++it) {
      int f = tid * 2 + it;                       // 0..511
      int krow = f >> 4, col = (f & 15) << 2;
      bReg[it] = *(const float4*)(B + (size_t)(k0 + krow) * N + n0 + col);
    }
  };
  auto commitB = [&](int buf) {
#pragma unroll
    for (int it = 0; it < 2; ++it) {
      int f = tid * 2 + it;
      int krow = f >> 4, col = (f & 15) << 2;
      Bts[buf][col + 0][krow] = (_Float16)bReg[it].x;
      Bts[buf][col + 1][krow] = (_Float16)bReg[it].y;
      Bts[buf][col + 2][krow] = (_Float16)bReg[it].z;
      Bts[buf][col + 3][krow] = (_Float16)bReg[it].w;
    }
  };

  // ---- prologue: stage buffer 0 ----
  issueA(0, 0);
  issueB(0);
  commitA(0);
  commitB(0);
  waitA();
  __syncthreads();

  // ---- pipelined main loop ----
  for (int ks = 0; ks < nk; ++ks) {
    int cur = ks & 1, nxt = cur ^ 1;
    bool more = (ks + 1 < nk);
    if (more) {                       // stage k+1 while computing k
      issueA(nxt, (ks + 1) * BK);
      issueB((ks + 1) * BK);
    }

    // A fragment (shared by 4 N-subtiles): lanes 0-15 K{0..7,16..23}, 16-31 K{8..15,24..31}
    int arow = wv * 16 + l15;
    int akb = lhalf ? 8 : 0;
    v8h a_lo = *(const v8h*)&As[cur][arow][akb];
    v8h a_hi = *(const v8h*)&As[cur][arow][akb + 16];
    v16h af;
#pragma unroll
    for (int j = 0; j < 8; ++j) { af[j] = a_lo[j]; af[j + 8] = a_hi[j]; }

    // 4 B fragments + 4 WMMAs: lanes 0-15 K{0..7,16..23}, 16-31 K{8..15,24..31}
    int bkb = lhalf ? 8 : 0;
#pragma unroll
    for (int ns = 0; ns < 4; ++ns) {
      v8h b_lo = *(const v8h*)&Bts[cur][ns * 16 + l15][bkb];
      v8h b_hi = *(const v8h*)&Bts[cur][ns * 16 + l15][bkb + 16];
      v16h bf;
#pragma unroll
      for (int j = 0; j < 8; ++j) { bf[j] = b_lo[j]; bf[j + 8] = b_hi[j]; }
      acc[ns] = __builtin_amdgcn_wmma_f32_16x16x32_f16(false, af, false, bf,
                                                       (short)0, acc[ns], false, false);
    }

    if (more) {
      commitA(nxt);
      commitB(nxt);
      waitA();
    }
    __syncthreads();
  }

  // ---- epilogue: optional C store (f16, +bias,+relu) and fused attention dot ----
  int rbase = m0 + wv * 16 + (lhalf ? 8 : 0);
  float attP[8] = {0.f, 0.f, 0.f, 0.f, 0.f, 0.f, 0.f, 0.f};
#pragma unroll
  for (int ns = 0; ns < 4; ++ns) {
    int col = n0 + ns * 16 + l15;
    float bv = bias ? bias[col] : 0.f;
    float av = attV ? attV[col] : 0.f;
#pragma unroll
    for (int r = 0; r < 8; ++r) {
      int row = rbase + r;
      if (C && row < M) {
        float v = acc[ns][r] + bv;
        if (relu) v = v > 0.f ? v : 0.f;
        C[(size_t)row * N + col] = (_Float16)v;
      }
      attP[r] += acc[ns][r] * av;
    }
  }
  if (attV) {
#pragma unroll
    for (int m = 1; m < 16; m <<= 1)
#pragma unroll
      for (int r = 0; r < 8; ++r)
        attP[r] += __shfl_xor(attP[r], m, 32);
    if (l15 == 0) {
      int h = nb;                            // BN==CPH: block nb covers head nb
#pragma unroll
      for (int r = 0; r < 8; ++r) {
        int row = rbase + r;
        if (row < M) attE[row * HEADS + h] = attP[r];
      }
    }
  }
}

// ---------- edge pass 1: cache logits + segment max via monotone-key atomicMax ----------
__global__ void k_edge_max(const int* __restrict__ src, const int* __restrict__ dst,
                           const float* __restrict__ ES, const float* __restrict__ ED,
                           float* __restrict__ eBuf, unsigned* __restrict__ mkey, int E)
{
  int t = blockIdx.x * blockDim.x + threadIdx.x;
  if (t >= E * HEADS) return;
  int e = t >> 2, h = t & 3;
  int s = src[e], d = dst[e];
  float v = leaky(ES[s * 4 + h] + ED[d * 4 + h]);
  eBuf[t] = v;
  atomicMax(mkey + d * 4 + h, f32key(v));
}

// ---------- edge pass 2: denom = segment sum of exp(e - m) ----------
__global__ void k_edge_sum(const int* __restrict__ dst, const float* __restrict__ eBuf,
                           const unsigned* __restrict__ mkey, float* __restrict__ denom, int E)
{
  int t = blockIdx.x * blockDim.x + threadIdx.x;
  if (t >= E * HEADS) return;
  int e = t >> 2, h = t & 3;
  int d = dst[e];
  atomicAdd(denom + d * 4 + h, expf(eBuf[t] - keyf32(mkey[d * 4 + h])));
}

// ---------- edge pass 3: wave per edge, scatter alpha * hs[src] (f16) into ACC[dst] ----------
__global__ __launch_bounds__(256) void k_edge_msg(
    const int* __restrict__ src, const int* __restrict__ dst,
    const float* __restrict__ eBuf,
    const unsigned* __restrict__ mkey, const float* __restrict__ denom,
    const _Float16* __restrict__ HS, float* __restrict__ ACC, int E)
{
  int gw = (blockIdx.x * blockDim.x + threadIdx.x) >> 5;
  int lane = threadIdx.x & 31;
  if (gw >= E) return;
  int s = src[gw], d = dst[gw];
  int h = lane >> 3;                           // 8 lanes per head (8 floats each)
  float v = eBuf[gw * 4 + h];
  float alpha = expf(v - keyf32(mkey[d * 4 + h])) / denom[d * 4 + h];
  v8h hx = *(const v8h*)(HS + (size_t)s * HID + lane * 8);   // 16B gather
  float* out = ACC + (size_t)d * HID + lane * 8;
#pragma unroll
  for (int j = 0; j < 8; ++j)
    atomicAdd(out + j, (float)hx[j] * alpha);
}

// ---------- finalize: OUT(f16) = elu(ACC + b1 (+ b2)) ----------
__global__ void k_finalize(const float* __restrict__ ACC, const float* __restrict__ b1,
                           const float* __restrict__ b2, _Float16* __restrict__ OUT, long long n)
{
  long long i = (long long)blockIdx.x * blockDim.x + threadIdx.x;
  if (i >= n) return;
  int col = (int)(i & (HID - 1));
  float v = ACC[i] + b1[col] + (b2 ? b2[col] : 0.f);
  OUT[i] = (_Float16)(v > 0.f ? v : expm1f(v));
}

// ---------- classifier: OUT[M,NCLS](f32) = X[M,HID](f16) @ W[HID,NCLS] + b ----------
__global__ void k_classifier(const _Float16* __restrict__ X, const float* __restrict__ W,
                             const float* __restrict__ b, float* __restrict__ OUT,
                             int M, int Ncls)
{
  int t = blockIdx.x * blockDim.x + threadIdx.x;
  if (t >= M * Ncls) return;
  int row = t / Ncls, col = t % Ncls;
  const v8h* xp8 = (const v8h*)(X + (size_t)row * HID);
  float s = 0.f;
#pragma unroll 2
  for (int k8 = 0; k8 < HID / 8; ++k8) {
    v8h x = xp8[k8];
    const float* w = W + (size_t)(k8 * 8) * Ncls + col;
#pragma unroll
    for (int j = 0; j < 8; ++j) s += (float)x[j] * w[(size_t)j * Ncls];
  }
  OUT[t] = s + b[col];
}

// =======================================================================
extern "C" void kernel_launch(void* const* d_in, const int* in_sizes, int n_in,
                              void* d_out, int out_size, void* d_ws, size_t ws_size,
                              hipStream_t stream)
{
  const float* x_paper  = (const float*)d_in[0];
  const float* x_author = (const float*)d_in[1];
  const int*   ei_w     = (const int*)d_in[2];
  const int*   ei_wr    = (const int*)d_in[3];
  const int*   ei_c     = (const int*)d_in[4];
  const float* Wp_in    = (const float*)d_in[5];
  const float* bp_in    = (const float*)d_in[6];
  const float* Wa_in    = (const float*)d_in[7];
  const float* ba_in    = (const float*)d_in[8];
  // GAT param groups (Ws,Wd,aS,aD,b) at 9,14,19,24,29,34 = l1w,l1wr,l1c,l2w,l2wr,l2c
  const float* Wcls = (const float*)d_in[39];
  const float* bcls = (const float*)d_in[40];

  const int Nn    = in_sizes[0] / IN_DIM_;  // 50000
  const int E     = in_sizes[2] / 2;        // 400000
  const int NCLS  = in_sizes[40];           // 349
  const int BATCH = out_size / NCLS;        // 1024
  const long long NH = (long long)Nn * HID;

  // ---- workspace layout (region stride = NH floats; f16 buffers use half) ----
  float* w = (float*)d_ws;
  _Float16* xp = (_Float16*)w; w += NH;
  _Float16* xa = (_Float16*)w; w += NH;
  _Float16* hs = (_Float16*)w; w += NH;
  float*    acc = w;           w += NH;
  _Float16* p1 = (_Float16*)w; w += NH;
  _Float16* a1 = (_Float16*)w; w += NH;
  float* es = w;               w += (size_t)Nn * HEADS;
  float* ed = w;               w += (size_t)Nn * HEADS;
  unsigned* mkey = (unsigned*)w; w += (size_t)Nn * HEADS;
  float* denom = w;            w += (size_t)Nn * HEADS;
  float* eBuf = w;             w += (size_t)E * HEADS;
  _Float16* p2 = xp;   // xp dead after layer 1

  auto zero = [&](void* ptr, long long n) {
    k_zero<<<(unsigned)((n + 255) / 256), 256, 0, stream>>>((float*)ptr, n);
  };
  const int gblocks = ((Nn + BM - 1) / BM) * (HID / BN);
  auto gemm_f32 = [&](const float* A, const float* B, _Float16* C, const float* bias,
                      int K, int relu) {
    k_wmma_gemm<float><<<gblocks, 256, 0, stream>>>(A, B, C, bias, nullptr, nullptr,
                                                    Nn, K, HID, relu);
  };
  auto gemm_f16 = [&](const _Float16* A, const float* B, _Float16* C,
                      const float* attV, float* attE) {
    k_wmma_gemm<_Float16><<<gblocks, 256, 0, stream>>>(A, B, C, nullptr, attV, attE,
                                                       Nn, HID, HID, 0);
  };
  // one GATConv: scatter-accumulate (without bias) into `acc`
  auto gat = [&](const _Float16* xsrc, const _Float16* xdst, const int* ei, int pbase) {
    const float* Ws = (const float*)d_in[pbase + 0];
    const float* Wd = (const float*)d_in[pbase + 1];
    const float* aS = (const float*)d_in[pbase + 2];
    const float* aD = (const float*)d_in[pbase + 3];
    gemm_f16(xsrc, Ws, hs, aS, es);            // hs + fused es
    gemm_f16(xdst, Wd, nullptr, aD, ed);       // ed only: hd is never materialized
    zero(mkey, (long long)Nn * HEADS);
    zero(denom, (long long)Nn * HEADS);
    const int* srcA = ei;
    const int* dstA = ei + E;
    int t4 = E * HEADS;
    k_edge_max<<<(t4 + 255) / 256, 256, 0, stream>>>(srcA, dstA, es, ed, eBuf, mkey, E);
    k_edge_sum<<<(t4 + 255) / 256, 256, 0, stream>>>(dstA, eBuf, mkey, denom, E);
    k_edge_msg<<<(E + 7) / 8, 256, 0, stream>>>(srcA, dstA, eBuf, mkey, denom, hs, acc, E);
  };
  auto fin = [&](int bidx1, int bidx2, _Float16* outbuf) {
    const float* b1 = (const float*)d_in[bidx1];
    const float* b2 = bidx2 >= 0 ? (const float*)d_in[bidx2] : nullptr;
    k_finalize<<<(unsigned)((NH + 255) / 256), 256, 0, stream>>>(acc, b1, b2, outbuf, NH);
  };

  // ---- input projections (relu) ----
  gemm_f32(x_paper,  Wp_in, xp, bp_in, IN_DIM_, 1);
  gemm_f32(x_author, Wa_in, xa, ba_in, IN_DIM_, 1);

  // ---- layer 1: paper dst (writes: author->paper) + (cites: paper->paper) ----
  zero(acc, NH);
  gat(xa, xp, ei_w, 9);    // l1w
  gat(xp, xp, ei_c, 19);   // l1c
  fin(13, 23, p1);

  // ---- layer 1: author dst (writes_rev: paper->author) ----
  zero(acc, NH);
  gat(xp, xa, ei_wr, 14);  // l1wr
  fin(18, -1, a1);

  // ---- layer 2: paper dst ----
  zero(acc, NH);
  gat(a1, p1, ei_w, 24);   // l2w
  gat(p1, p1, ei_c, 34);   // l2c
  fin(28, 38, p2);

  // ---- classifier on first BATCH rows ----
  int tot = BATCH * NCLS;
  k_classifier<<<(tot + 255) / 256, 256, 0, stream>>>(p2, Wcls, bcls, (float*)d_out, BATCH, NCLS);
}
